// outconv_50122268344946
// MI455X (gfx1250) — matrix operand order repaired and checked by reading, not compiled, not run on hardware
//
#include <hip/hip_runtime.h>
#include <hip/hip_bf16.h>

// CDNA5 / gfx1250: wave32, WMMA bf16 path.
typedef __attribute__((ext_vector_type(16))) __bf16 v16bf;
typedef __attribute__((ext_vector_type(8)))  float  v8f;

#define Bn    16
#define CIN_  64
#define COUT_ 4
#define Fdim  257
#define Tdim  512
#define NPB   (Fdim * Tdim)        // 131584 pixels per batch image
#define TILES_PER_B (NPB / 32)     // 4112
#define NTILES (Bn * TILES_PER_B)  // 65792
#define WAVES_PER_BLOCK 8
#define THREADS (WAVES_PER_BLOCK * 32)
#define NBLOCKS 2056               // 16448 waves -> exactly 4 tiles per wave

__global__ __launch_bounds__(THREADS)
void outconv_wmma_kernel(const float* __restrict__ xd,   // [B,64,F,T]
                         const float* __restrict__ xs,   // [B,4,F,T]
                         const float* __restrict__ W1,   // [11,64]
                         const float* __restrict__ b1,   // [11]
                         const float* __restrict__ W2,   // [11,15]
                         const float* __restrict__ b2,   // [11]
                         float* __restrict__ out)        // [B,4,F,T]
{
  __shared__ float hstage[WAVES_PER_BLOCK][32][12] __attribute__((aligned(16)));
  __shared__ float sW2[11 * 15];
  __shared__ float sb2[11];

  const int tid = threadIdx.x;
  if (tid < 165) sW2[tid] = W2[tid];
  if (tid < 11)  sb2[tid] = b2[tid];
  __syncthreads();

  const int lane = tid & 31;
  const int wv   = tid >> 5;
  const int half = lane >> 4;   // 0: lanes 0-15, 1: lanes 16-31
  const int l16  = lane & 15;

  // ---- Preload A = W1 in WMMA A-layout (bf16), rows >= 11 zero ----
  // lanes 0-15: elems 0..15 -> K 0..15
  // lanes 16-31: elems 0..15 -> K 16..31
  const int m = l16;            // A row = conv1 output channel
  v16bf A0, A1;
#pragma unroll
  for (int e = 0; e < 16; ++e) {
    int k = half * 16 + e;                              // K in 0..31
    float w0 = (m < 11) ? W1[m * CIN_ + k]       : 0.f;
    float w1 = (m < 11) ? W1[m * CIN_ + 32 + k]  : 0.f;
    A0[e] = (__bf16)w0;    // native cvt (RNE)
    A1[e] = (__bf16)w1;
  }

  // C-matrix rows held by this lane: rbase..rbase+7 ; seed with bias b1
  const int rbase = half * 8;
  v8f cinit;
#pragma unroll
  for (int v = 0; v < 8; ++v) {
    int row = rbase + v;
    cinit[v] = (row < 11) ? b1[row] : 0.f;
  }

  const int waveId = blockIdx.x * WAVES_PER_BLOCK + wv;
  const int nWaves = gridDim.x * WAVES_PER_BLOCK;

  for (int tile = waveId; tile < NTILES; tile += nWaves) {
    const int b  = tile / TILES_PER_B;
    const int p0 = (tile - b * TILES_PER_B) * 32;

    // ================= conv1 via WMMA: two groups of 16 pixels ============
#pragma unroll
    for (int g = 0; g < 2; ++g) {
      const int p     = p0 + g * 16 + l16;   // this lane's B-column pixel
      const int kbase = half * 16;           // B layout: lane half picks K half
      v8f acc = cinit;

#pragma unroll
      for (int kh = 0; kh < 2; ++kh) {
        const float* xb = xd + (size_t)(b * CIN_ + kbase + kh * 32) * NPB + p;
        // 16 streaming loads (constant 526336-B immediate offsets), then
        // pairwise fp32->bf16 (compiler can fuse into v_cvt_pk_bf16_f32)
        float f[16];
#pragma unroll
        for (int e = 0; e < 16; ++e)
          f[e] = __builtin_nontemporal_load(&xb[(size_t)e * NPB]);
        v16bf Bm;
#pragma unroll
        for (int e = 0; e < 16; ++e) Bm[e] = (__bf16)f[e];
        acc = __builtin_amdgcn_wmma_f32_16x16x32_bf16(
                  false, (kh == 0) ? A0 : A1, false, Bm,
                  (short)0, acc, false, false);
      }
      // h = relu(W1 x + b1) -> per-wave LDS stage (rows split across halves)
#pragma unroll
      for (int v = 0; v < 8; ++v) {
        int row = rbase + v;
        if (row < 11) hstage[wv][g * 16 + l16][row] = fmaxf(acc[v], 0.f);
      }
    }
    // Same-wave DS ops are in-order; backend inserts s_wait_dscnt for the
    // RAW on hstage below. No cross-wave sharing -> no barrier needed.

    // ================= epilogue: one lane per pixel =======================
    {
      const int p = p0 + lane;
      const float xs0 = __builtin_nontemporal_load(&xs[(size_t)(b * COUT_ + 0) * NPB + p]);
      const float xs1 = __builtin_nontemporal_load(&xs[(size_t)(b * COUT_ + 1) * NPB + p]);
      const float xs2 = __builtin_nontemporal_load(&xs[(size_t)(b * COUT_ + 2) * NPB + p]);
      const float xs3 = __builtin_nontemporal_load(&xs[(size_t)(b * COUT_ + 3) * NPB + p]);

      const float4 h03 = *(const float4*)&hstage[wv][lane][0];
      const float4 h47 = *(const float4*)&hstage[wv][lane][4];
      const float4 h8a = *(const float4*)&hstage[wv][lane][8];
      const float h[11] = {h03.x, h03.y, h03.z, h03.w,
                           h47.x, h47.y, h47.z, h47.w,
                           h8a.x, h8a.y, h8a.z};

      float o2[11];
#pragma unroll
      for (int o = 0; o < 11; ++o) {
        const float* w = &sW2[o * 15];
        float s = sb2[o] + w[0] * xs0 + w[1] * xs1 + w[2] * xs2 + w[3] * xs3;
#pragma unroll
        for (int j = 0; j < 11; ++j) s += w[4 + j] * h[j];
        o2[o] = s;
      }

      float t[9];
#pragma unroll
      for (int i = 0; i < 9; ++i) t[i] = o2[i] / (1.f + fabsf(o2[i]));
      const float m0 = 1.f / (1.f + __expf(-o2[9]));
      const float m1 = 1.f / (1.f + __expf(-o2[10]));

      const float y0 = m0 * (t[0] * xs0 + t[1] * xs1 + t[2] * xs2);
      const float y1 = m0 * (t[3] * xs0 + t[4] * xs1 + t[5] * xs2);
      const float y2 = m0 * (t[6] * xs0 + t[7] * xs1 + t[8] * xs2);
      const float y3 = m1 * xs3;

      float* ob = out + (size_t)b * COUT_ * NPB + p;
      __builtin_nontemporal_store(y0, &ob[0 * NPB]);
      __builtin_nontemporal_store(y1, &ob[1 * NPB]);
      __builtin_nontemporal_store(y2, &ob[2 * NPB]);
      __builtin_nontemporal_store(y3, &ob[3 * NPB]);
    }
  }
}

extern "C" void kernel_launch(void* const* d_in, const int* in_sizes, int n_in,
                              void* d_out, int out_size, void* d_ws, size_t ws_size,
                              hipStream_t stream) {
  const float* xd = (const float*)d_in[0];  // x_decode [16,64,257,512]
  const float* xs = (const float*)d_in[1];  // x_skip   [16,4,257,512]
  const float* W1 = (const float*)d_in[2];  // [11,64]
  const float* b1 = (const float*)d_in[3];  // [11]
  const float* W2 = (const float*)d_in[4];  // [11,15]
  const float* b2 = (const float*)d_in[5];  // [11]
  float* out = (float*)d_out;               // [16,4,257,512]

  outconv_wmma_kernel<<<NBLOCKS, THREADS, 0, stream>>>(xd, xs, W1, b1, W2, b2, out);
}
